// TransformerEncoderMemEffSA_89532888252914
// MI455X (gfx1250) — matrix operand order repaired and checked by reading, not compiled, not run on hardware
//
#include <hip/hip_runtime.h>

typedef __attribute__((ext_vector_type(16))) _Float16 v16h;
typedef __attribute__((ext_vector_type(8)))  _Float16 v8h;
typedef __attribute__((ext_vector_type(8)))  float    v8f;
typedef __attribute__((ext_vector_type(4)))  unsigned int u32x4;
typedef __attribute__((ext_vector_type(8)))  int          i32x8;
typedef __attribute__((ext_vector_type(4)))  int          i32x4;

#define DEVI __device__ __forceinline__

#if __has_builtin(__builtin_amdgcn_tensor_load_to_lds) && \
    __has_builtin(__builtin_amdgcn_s_wait_tensorcnt)
#define USE_TDM 1
#else
#define USE_TDM 0
#endif

namespace {

constexpr int Bb = 4, Ss = 2048, Cc = 512, Hh = 8, Dd = 64;
constexpr int Mrows = Bb * Ss;          // 8192
constexpr float kScale = 0.125f;        // D^-0.5
constexpr float kEps = 1e-5f;

// ---- WMMA fragment loaders (wave32, 16x16x32 f16) ----
// A (16x32, MxK): lanes 0-15 hold row M=lane, K=0..7 & 16..23;
//                 lanes 16-31 hold row M=lane-16, K=8..15 & 24..31.
DEVI v16h frag_a(const _Float16* p, int ld) {
  const int lane = threadIdx.x & 31;
  const _Float16* q = p + (lane & 15) * ld + ((lane >> 4) * 8);
  v8h lo = *(const v8h*)(q);
  v8h hi = *(const v8h*)(q + 16);
  v16h out;
#pragma unroll
  for (int i = 0; i < 8; ++i) { out[i] = lo[i]; out[i + 8] = hi[i]; }
  return out;
}

// B (32x16, KxN) loaded from a K-major ("Bt[n][k]") LDS tile:
// lanes 0-15 hold col N=lane, K=0..7 & 16..23; lanes 16-31 hold col N=lane-16, K=8..15 & 24..31.
DEVI v16h frag_b(const _Float16* p, int ld) {
  const int lane = threadIdx.x & 31;
  const _Float16* q = p + (lane & 15) * ld + ((lane >> 4) * 8);
  v8h lo = *(const v8h*)(q);
  v8h hi = *(const v8h*)(q + 16);
  v16h out;
#pragma unroll
  for (int i = 0; i < 8; ++i) { out[i] = lo[i]; out[i + 8] = hi[i]; }
  return out;
}

#if USE_TDM
// Generic LDS pointers carry the LDS byte offset in addr[31:0] (ISA 10.2).
DEVI unsigned lds_off(const void* p) {
  return (unsigned)(unsigned long long)(uintptr_t)p;
}

// Issue a 2D TDM tile load: tile_d1 rows x tile_d0 elements (f16),
// row stride `stride` elements, packed contiguously into LDS at lds_addr.
// D# layout per CDNA5 ISA.
DEVI void tdm_load_2d_f16(unsigned lds_addr, const void* gptr,
                          unsigned tensor_d0, unsigned tensor_d1,
                          unsigned tile_d0, unsigned tile_d1,
                          unsigned stride_elems) {
  unsigned long long ga = (unsigned long long)(uintptr_t)gptr;
  u32x4 g0;
  g0[0] = 1u;                                             // count=1, user desc
  g0[1] = lds_addr;                                       // lds_addr[31:0]
  g0[2] = (unsigned)ga;                                   // global_addr[31:0]
  g0[3] = (unsigned)((ga >> 32) & 0x1FFFFFFu) | (2u << 30);  // addr[56:32]|type=2
  i32x8 g1;
  g1[0] = (int)(1u << 16);                                // data_size=1 (2 bytes)
  g1[1] = (int)((tensor_d0 & 0xFFFFu) << 16);             // tensor_dim0[15:0]
  g1[2] = (int)(((tensor_d0 >> 16) & 0xFFFFu) |
                ((tensor_d1 & 0xFFFFu) << 16));           // dim0 hi | dim1 lo
  g1[3] = (int)(((tensor_d1 >> 16) & 0xFFFFu) |
                ((tile_d0 & 0xFFFFu) << 16));             // dim1 hi | tile_dim0
  g1[4] = (int)(tile_d1 & 0xFFFFu);                       // tile_dim1, tile_dim2=0
  g1[5] = (int)stride_elems;                              // tensor_dim0_stride lo
  g1[6] = 0;                                              // stride hi | dim1_stride lo
  g1[7] = 0;
  i32x4 z4 = {0, 0, 0, 0};
#if defined(__clang_major__) && (__clang_major__ >= 23)
  i32x8 z8 = {0, 0, 0, 0, 0, 0, 0, 0};
  __builtin_amdgcn_tensor_load_to_lds(g0, g1, z4, z4, z8, 0);
#else
  __builtin_amdgcn_tensor_load_to_lds(g0, g1, z4, z4, 0);
#endif
}
#endif  // USE_TDM

// ---- prep kernels ----
__global__ __launch_bounds__(256) void cast_f16_kernel(const float* __restrict__ in,
                                                       _Float16* __restrict__ out, int n) {
  int i = blockIdx.x * 256 + threadIdx.x;
  if (i < n) out[i] = (_Float16)in[i];
}

// W [K=512, N=512] row-major fp32 -> Wt [N, K] f16
__global__ __launch_bounds__(256) void transpose_cast_kernel(const float* __restrict__ in,
                                                             _Float16* __restrict__ out) {
  int i = blockIdx.x * 256 + threadIdx.x;  // i = k*512 + n
  int k = i >> 9, n = i & 511;
  out[(size_t)n * 512 + k] = (_Float16)in[i];
}

// ---- tiled GEMM: C[M,N] = A[M,K](f16) * Bt[N,K](f16), f32 accum ----
// TDM double-buffered: wave 0 issues next A/B tile DMA, waits tensorcnt<=2
// so the current pair is resident, then all 8 waves run WMMA on it.
template <bool OUT16, bool BIAS>
__global__ __launch_bounds__(256) void gemm_f16(
    const _Float16* __restrict__ A, const _Float16* __restrict__ Bt,
    const float* __restrict__ bias, float* __restrict__ Cf,
    _Float16* __restrict__ Ch, int Kd, int Nd) {
  __shared__ _Float16 sA[2][128 * 32];
  __shared__ _Float16 sB[2][64 * 32];
  const int tid = threadIdx.x;
  const int lane = tid & 31;
  const int wave = tid >> 5;       // 0..7
  const int wm = wave & 3;         // M sub-tile
  const int wn = wave >> 2;        // N sub-tile
  const int m0 = blockIdx.x * 128;
  const int n0 = blockIdx.y * 64;

  v8f acc[2][2] = {};
  const int nk = Kd / 32;

#if USE_TDM
  if (wave == 0) {
    tdm_load_2d_f16(lds_off(&sA[0][0]), A + (size_t)m0 * Kd, Kd, 128, 32, 128, Kd);
    tdm_load_2d_f16(lds_off(&sB[0][0]), Bt + (size_t)n0 * Kd, Kd, 64, 32, 64, Kd);
  }
#else
  const int lr = tid >> 2;         // 0..63
  const int lc = (tid & 3) * 8;    // 8-half (16B) column chunk
#endif

  for (int i = 0; i < nk; ++i) {
    const int cur = i & 1;
#if USE_TDM
    if (wave == 0) {
      if (i + 1 < nk) {
        const int k1 = (i + 1) * 32;
        tdm_load_2d_f16(lds_off(&sA[cur ^ 1][0]), A + (size_t)m0 * Kd + k1,
                        Kd, 128, 32, 128, Kd);
        tdm_load_2d_f16(lds_off(&sB[cur ^ 1][0]), Bt + (size_t)n0 * Kd + k1,
                        Kd, 64, 32, 64, Kd);
        __builtin_amdgcn_s_wait_tensorcnt(2);  // in-order: pair i complete
      } else {
        __builtin_amdgcn_s_wait_tensorcnt(0);
      }
    }
    __syncthreads();
#else
    {
      const int k0 = i * 32;
      *(v8h*)&sA[cur][lr * 32 + lc]        = *(const v8h*)&A[(size_t)(m0 + lr) * Kd + k0 + lc];
      *(v8h*)&sA[cur][(lr + 64) * 32 + lc] = *(const v8h*)&A[(size_t)(m0 + lr + 64) * Kd + k0 + lc];
      *(v8h*)&sB[cur][lr * 32 + lc]        = *(const v8h*)&Bt[(size_t)(n0 + lr) * Kd + k0 + lc];
      __syncthreads();
    }
#endif

    v16h af[2], bf[2];
#pragma unroll
    for (int ii = 0; ii < 2; ++ii) af[ii] = frag_a(&sA[cur][(wm * 32 + ii * 16) * 32], 32);
#pragma unroll
    for (int j = 0; j < 2; ++j) bf[j] = frag_b(&sB[cur][(wn * 32 + j * 16) * 32], 32);
#pragma unroll
    for (int ii = 0; ii < 2; ++ii)
#pragma unroll
      for (int j = 0; j < 2; ++j)
        acc[ii][j] = __builtin_amdgcn_wmma_f32_16x16x32_f16(
            false, af[ii], false, bf[j], (short)0, acc[ii][j], false, false);
    __syncthreads();
  }

  const int colSub = lane & 15;
  const int hi8 = (lane >> 4) * 8;
#pragma unroll
  for (int ii = 0; ii < 2; ++ii)
#pragma unroll
    for (int j = 0; j < 2; ++j) {
      int nn = n0 + wn * 32 + j * 16 + colSub;
      float bv = BIAS ? bias[nn] : 0.f;
#pragma unroll
      for (int r = 0; r < 8; ++r) {
        int mm = m0 + wm * 32 + ii * 16 + hi8 + r;
        float v = acc[ii][j][r] + bv;
        if (OUT16) Ch[(size_t)mm * Nd + nn] = (_Float16)v;
        else       Cf[(size_t)mm * Nd + nn] = v;
      }
    }
}

// ---- LayerNorm: one wave per 512-wide row ----
template <bool RES, bool OUT32, bool OUT16>
__global__ __launch_bounds__(128) void ln_kernel(
    const float* __restrict__ x, const float* __restrict__ res,
    const float* __restrict__ w, const float* __restrict__ b,
    float* __restrict__ of, _Float16* __restrict__ oh) {
  const int lane = threadIdx.x & 31;
  const int wv = threadIdx.x >> 5;
  const size_t row = (size_t)blockIdx.x * 4 + wv;
  const float* xr = x + row * Cc;
  float vals[16];
  float s = 0.f;
#pragma unroll
  for (int i = 0; i < 16; ++i) {
    int c = lane + i * 32;
    float v = xr[c];
    if (RES) v += res[row * Cc + c];
    vals[i] = v;
    s += v;
  }
#pragma unroll
  for (int msk = 16; msk >= 1; msk >>= 1) s += __shfl_xor(s, msk, 32);
  const float mean = s * (1.f / Cc);
  float vs = 0.f;
#pragma unroll
  for (int i = 0; i < 16; ++i) { float d = vals[i] - mean; vs += d * d; }
#pragma unroll
  for (int msk = 16; msk >= 1; msk >>= 1) vs += __shfl_xor(vs, msk, 32);
  const float inv = rsqrtf(vs * (1.f / Cc) + kEps);
#pragma unroll
  for (int i = 0; i < 16; ++i) {
    int c = lane + i * 32;
    float y = (vals[i] - mean) * inv * w[c] + b[c];
    if (OUT32) of[row * Cc + c] = y;
    if (OUT16) oh[row * Cc + c] = (_Float16)y;
  }
}

// ---- flash-style causal attention ----
// grid: (S/64, H, B); block: 128 (4 waves), each wave owns 16 query rows.
__global__ __launch_bounds__(128) void attn_kernel(
    const _Float16* __restrict__ Q, const _Float16* __restrict__ K,
    const _Float16* __restrict__ V, _Float16* __restrict__ O) {
  __shared__ _Float16 sK[64 * 64];        // [key][d]  == Bt layout for Q*K^T
  __shared__ _Float16 sVt[64 * 64];       // [d][key]  == Bt layout for P*V
  __shared__ _Float16 sP[4 * 16 * 64];    // per-wave P tile

  const int tid = threadIdx.x;
  const int lane = tid & 31;
  const int w = tid >> 5;
  const int qb = blockIdx.x * 64;
  const int h = blockIdx.y;
  const int b = blockIdx.z;
  const size_t rowBase = (size_t)b * Ss;
  const int hOff = h * Dd;
  const int colSub = lane & 15;
  const int hi8 = (lane >> 4) * 8;

  // Q fragments for this wave's 16 rows (two K=32 chunks over D=64)
  v16h qf[2];
  {
    const _Float16* qp = Q + (rowBase + qb + w * 16 + (lane & 15)) * (size_t)Cc
                         + hOff + ((lane >> 4) * 8);
#pragma unroll
    for (int ch = 0; ch < 2; ++ch) {
      v8h lo = *(const v8h*)(qp + ch * 32);
      v8h hi = *(const v8h*)(qp + ch * 32 + 16);
#pragma unroll
      for (int i = 0; i < 8; ++i) { qf[ch][i] = lo[i]; qf[ch][i + 8] = hi[i]; }
    }
  }

  float mrow[8], lrow[8];
  v8f oacc[4] = {};
#pragma unroll
  for (int r = 0; r < 8; ++r) { mrow[r] = -1e30f; lrow[r] = 0.f; }

  _Float16* sPw = &sP[w * 16 * 64];

  for (int kb = 0; kb <= qb; kb += 64) {
    // stage K block [64 keys][64 d] and V block transposed [64 d][64 keys]
    {
      int r = tid >> 1;
      int c0 = (tid & 1) * 32;
      const _Float16* ksrc = K + (rowBase + kb + r) * (size_t)Cc + hOff + c0;
#pragma unroll
      for (int u = 0; u < 4; ++u)
        *(v8h*)&sK[r * 64 + c0 + u * 8] = *(const v8h*)(ksrc + u * 8);
      const _Float16* vsrc = V + (rowBase + kb + r) * (size_t)Cc + hOff + c0;
      v8h tmp[4];
#pragma unroll
      for (int u = 0; u < 4; ++u) tmp[u] = *(const v8h*)(vsrc + u * 8);
#pragma unroll
      for (int u = 0; u < 32; ++u) sVt[(c0 + u) * 64 + r] = tmp[u >> 3][u & 7];
    }
    __syncthreads();

    // scores S = Q * K^T for 4 key sub-tiles of 16
    v8f sfr[4];
    float bmax[8];
#pragma unroll
    for (int r = 0; r < 8; ++r) bmax[r] = -1e30f;
#pragma unroll
    for (int st = 0; st < 4; ++st) {
      v8f s = {};
#pragma unroll
      for (int ch = 0; ch < 2; ++ch) {
        v16h kf = frag_b(&sK[st * 16 * 64 + ch * 32], 64);
        s = __builtin_amdgcn_wmma_f32_16x16x32_f16(false, qf[ch], false, kf,
                                                   (short)0, s, false, false);
      }
      int kcol = kb + st * 16 + colSub;
#pragma unroll
      for (int r = 0; r < 8; ++r) {
        int qrow = qb + w * 16 + hi8 + r;
        float v = (kcol > qrow) ? -1e30f : s[r] * kScale;
        s[r] = v;
        bmax[r] = fmaxf(bmax[r], v);
      }
      sfr[st] = s;
    }
    // row-max across the 16 lanes holding this row's 16 columns
#pragma unroll
    for (int r = 0; r < 8; ++r)
#pragma unroll
      for (int msk = 8; msk >= 1; msk >>= 1)
        bmax[r] = fmaxf(bmax[r], __shfl_xor(bmax[r], msk, 32));

    float corr[8], lsum[8];
#pragma unroll
    for (int r = 0; r < 8; ++r) {
      float mn = fmaxf(mrow[r], bmax[r]);
      corr[r] = __expf(mrow[r] - mn);
      mrow[r] = mn;
      lsum[r] = 0.f;
    }
    // P = exp(S - m); spill C-layout -> LDS -> reload in A layout
#pragma unroll
    for (int st = 0; st < 4; ++st) {
      v8f s = sfr[st];
#pragma unroll
      for (int r = 0; r < 8; ++r) {
        float pv = __expf(s[r] - mrow[r]);
        lsum[r] += pv;
        sPw[(hi8 + r) * 64 + st * 16 + colSub] = (_Float16)pv;
      }
    }
#pragma unroll
    for (int r = 0; r < 8; ++r) {
#pragma unroll
      for (int msk = 8; msk >= 1; msk >>= 1) lsum[r] += __shfl_xor(lsum[r], msk, 32);
      lrow[r] = lrow[r] * corr[r] + lsum[r];
    }
#pragma unroll
    for (int j = 0; j < 4; ++j)
#pragma unroll
      for (int r = 0; r < 8; ++r) oacc[j][r] = oacc[j][r] * corr[r];

    v16h pf[2];
#pragma unroll
    for (int ch = 0; ch < 2; ++ch) pf[ch] = frag_a(&sPw[ch * 32], 64);
#pragma unroll
    for (int j = 0; j < 4; ++j)
#pragma unroll
      for (int ch = 0; ch < 2; ++ch) {
        v16h vf = frag_b(&sVt[j * 16 * 64 + ch * 32], 64);
        oacc[j] = __builtin_amdgcn_wmma_f32_16x16x32_f16(false, pf[ch], false, vf,
                                                         (short)0, oacc[j], false, false);
      }
    __syncthreads();
  }

#pragma unroll
  for (int j = 0; j < 4; ++j)
#pragma unroll
    for (int r = 0; r < 8; ++r) {
      int qrow = qb + w * 16 + hi8 + r;
      float v = oacc[j][r] / lrow[r];
      O[(rowBase + qrow) * (size_t)Cc + hOff + j * 16 + colSub] = (_Float16)v;
    }
}

}  // namespace

extern "C" void kernel_launch(void* const* d_in, const int* in_sizes, int n_in,
                              void* d_out, int out_size, void* d_ws, size_t ws_size,
                              hipStream_t stream) {
  (void)in_sizes; (void)n_in; (void)out_size; (void)ws_size;
  const float* x      = (const float*)d_in[0];
  const float* Wq     = (const float*)d_in[1];
  const float* Wk     = (const float*)d_in[2];
  const float* Wv     = (const float*)d_in[3];
  const float* Wo     = (const float*)d_in[4];
  const float* ln1_w  = (const float*)d_in[5];
  const float* ln1_b  = (const float*)d_in[6];
  const float* ffl1_w = (const float*)d_in[7];
  const float* ffl1_b = (const float*)d_in[8];
  const float* ff_w1  = (const float*)d_in[9];
  const float* ff_b1  = (const float*)d_in[10];
  const float* ffl2_w = (const float*)d_in[11];
  const float* ffl2_b = (const float*)d_in[12];
  const float* ff_w2  = (const float*)d_in[13];
  const float* ff_b2  = (const float*)d_in[14];
  const float* ln2_w  = (const float*)d_in[15];
  const float* ln2_b  = (const float*)d_in[16];
  float* out = (float*)d_out;

  char* p = (char*)d_ws;
  auto carve = [&](size_t bytes) {
    void* r = (void*)p;
    p += (bytes + 255) & ~(size_t)255;
    return r;
  };
  const size_t nTok = (size_t)Mrows * Cc;  // 8192*512
  _Float16* xh   = (_Float16*)carve(nTok * 2);
  _Float16* wqt  = (_Float16*)carve((size_t)Cc * Cc * 2);
  _Float16* wkt  = (_Float16*)carve((size_t)Cc * Cc * 2);
  _Float16* wvt  = (_Float16*)carve((size_t)Cc * Cc * 2);
  _Float16* wot  = (_Float16*)carve((size_t)Cc * Cc * 2);
  _Float16* w1t  = (_Float16*)carve((size_t)Cc * Cc * 2);
  _Float16* w2t  = (_Float16*)carve((size_t)Cc * Cc * 2);
  _Float16* Qh   = (_Float16*)carve(nTok * 2);
  _Float16* Kh   = (_Float16*)carve(nTok * 2);
  _Float16* Vh   = (_Float16*)carve(nTok * 2);
  _Float16* Ah   = (_Float16*)carve(nTok * 2);   // attention output (f16)
  float*    bufA = (float*)   carve(nTok * 4);   // a_out -> h -> h2 (reused)
  float*    xln  = (float*)   carve(nTok * 4);   // x_ln (f32, kept for final residual)
  _Float16* th   = (_Float16*)carve(nTok * 2);   // LN'd f16 GEMM input (reused)

  const dim3 gGemm(Mrows / 128, Cc / 64);
  const int gLN = Mrows / 4;

  // prep: cast x, transpose+cast weights
  cast_f16_kernel<<<(int)(nTok / 256), 256, 0, stream>>>(x, xh, (int)nTok);
  transpose_cast_kernel<<<1024, 256, 0, stream>>>(Wq, wqt);
  transpose_cast_kernel<<<1024, 256, 0, stream>>>(Wk, wkt);
  transpose_cast_kernel<<<1024, 256, 0, stream>>>(Wv, wvt);
  transpose_cast_kernel<<<1024, 256, 0, stream>>>(Wo, wot);
  transpose_cast_kernel<<<1024, 256, 0, stream>>>(ff_w1, w1t);
  transpose_cast_kernel<<<1024, 256, 0, stream>>>(ff_w2, w2t);

  // Q, K, V projections (f16 out)
  gemm_f16<true, false><<<gGemm, 256, 0, stream>>>(xh, wqt, nullptr, nullptr, Qh, Cc, Cc);
  gemm_f16<true, false><<<gGemm, 256, 0, stream>>>(xh, wkt, nullptr, nullptr, Kh, Cc, Cc);
  gemm_f16<true, false><<<gGemm, 256, 0, stream>>>(xh, wvt, nullptr, nullptr, Vh, Cc, Cc);

  // causal attention
  attn_kernel<<<dim3(Ss / 64, Hh, Bb), 128, 0, stream>>>(Qh, Kh, Vh, Ah);

  // a = attn @ Wo (f32); x_ln = LN(a + x)
  gemm_f16<false, false><<<gGemm, 256, 0, stream>>>(Ah, wot, nullptr, bufA, nullptr, Cc, Cc);
  ln_kernel<true, true, false><<<gLN, 128, 0, stream>>>(bufA, x, ln1_w, ln1_b, xln, nullptr);

  // h = LN(x_ln, ff_ln1) @ ff_w1 + ff_b1
  ln_kernel<false, false, true><<<gLN, 128, 0, stream>>>(xln, nullptr, ffl1_w, ffl1_b, nullptr, th);
  gemm_f16<false, true><<<gGemm, 256, 0, stream>>>(th, w1t, ff_b1, bufA, nullptr, Cc, Cc);

  // h2 = LN(h, ff_ln2) @ ff_w2 + ff_b2
  ln_kernel<false, false, true><<<gLN, 128, 0, stream>>>(bufA, nullptr, ffl2_w, ffl2_b, nullptr, th);
  gemm_f16<false, true><<<gGemm, 256, 0, stream>>>(th, w2t, ff_b2, bufA, nullptr, Cc, Cc);

  // out = LN(h2 + x_ln, ln2)
  ln_kernel<true, true, false><<<gLN, 128, 0, stream>>>(bufA, xln, ln2_w, ln2_b, out, nullptr);
}
